// GAT_61512521613452
// MI455X (gfx1250) — compile-verified, matrix operand order checked by reading, not run on hardware
//
#include <hip/hip_runtime.h>
#include <stdint.h>

// ---------------------------------------------------------------------------
// Types for CDNA5 WMMA (wave32): A/B fragments are 16 bf16 per lane (8 VGPRs),
// C/D is 8 f32 per lane.
// ---------------------------------------------------------------------------
typedef __attribute__((ext_vector_type(16))) __bf16 v16bf;
typedef __attribute__((ext_vector_type(8)))  float  v8f;

struct U128 { unsigned x, y, z, w; };
struct U256 { U128 lo, hi; };

// ---------------------------------------------------------------------------
// Utility kernels
// ---------------------------------------------------------------------------
__global__ void fill_u32_kernel(unsigned* __restrict__ p, unsigned val, size_t n) {
  size_t i = (size_t)blockIdx.x * blockDim.x + threadIdx.x;
  if (i < n) p[i] = val;
}

// float -> bf16 (RNE) with row/col zero padding: src is [srcRows, kin],
// dst is [rowsOut, kout] (rowsOut implied by total = rowsOut*kout).
__global__ void f2bf_pad_kernel(const float* __restrict__ src,
                                unsigned short* __restrict__ dst,
                                int srcRows, int kin, int kout, size_t total) {
  size_t i = (size_t)blockIdx.x * blockDim.x + threadIdx.x;
  if (i >= total) return;
  size_t row = i / (size_t)kout;
  int c = (int)(i - row * (size_t)kout);
  float v = 0.0f;
  if (row < (size_t)srcRows && c < kin) v = src[row * (size_t)kin + c];
  unsigned u = __float_as_uint(v);
  unsigned r = u + 0x7FFFu + ((u >> 16) & 1u);   // round-to-nearest-even
  dst[i] = (unsigned short)(r >> 16);
}

// ---------------------------------------------------------------------------
// bf16 WMMA GEMM:  C[M, Np] = A[M, K] * B[Np, K]^T   (both row-major, K-contig)
// Wave-level blocking: each wave computes a 16x64 strip (1 A-fragment reused
// across 4 N-tiles -> 4 accumulators). 8 waves per block cover 8 M-tiles of
// the same 4 N-tiles, so B fragments are shared through L1/L2.
// Fragment layout per ISA 7.12.2 (16-bit A 16x32 / B 32x16):
//   lane L (half = L>>4, r = L&15): elements 0..7  = K[half*8 .. half*8+7]
//                                   elements 8..15 = K[16+half*8 .. 16+half*8+7]
// Requires Np % 64 == 0 (1024 and 768 both qualify).
// ---------------------------------------------------------------------------
__device__ __forceinline__ v16bf load_frag(const unsigned short* p) {
  U256 t;
  t.lo = *(const U128*)(p);
  t.hi = *(const U128*)(p + 16);
  return __builtin_bit_cast(v16bf, t);
}

__global__ void gemm_bf16_wmma_kernel(const unsigned short* __restrict__ A,
                                      const unsigned short* __restrict__ B,
                                      float* __restrict__ C,
                                      int M, int Np, int K, int ldc) {
  const int wave = threadIdx.x >> 5;
  const int lane = threadIdx.x & 31;
  const int tm  = blockIdx.y * 8 + wave;     // M-tile of this wave
  const int tn0 = blockIdx.x * 4;            // first of 4 N-tiles
  if (tm * 16 >= M) return;                  // wave-uniform: EXEC stays all-1s
  const int half = lane >> 4;
  const int r16  = lane & 15;

  const unsigned short* arow = A + (size_t)(tm * 16 + r16) * K + half * 8;
  const unsigned short* brow = B + (size_t)(tn0 * 16 + r16) * K + half * 8;
  const size_t bts = (size_t)16 * K;         // B row stride between N-tiles

  v8f acc[4] = {v8f{}, v8f{}, v8f{}, v8f{}};
  for (int k = 0; k < K; k += 32) {
    __builtin_prefetch(arow + k + 128, 0, 1);
    __builtin_prefetch(brow + k + 128, 0, 1);
    v16bf av = load_frag(arow + k);
    v16bf bv[4];
#pragma unroll
    for (int t = 0; t < 4; ++t) bv[t] = load_frag(brow + bts * t + k);
#pragma unroll
    for (int t = 0; t < 4; ++t) {
      // (neg_a, A, neg_b, B, c_mod, C, reuse_a, reuse_b)
      acc[t] = __builtin_amdgcn_wmma_f32_16x16x32_bf16(false, av, false, bv[t],
                                                       (short)0, acc[t],
                                                       false, false);
    }
  }

  // C/D layout: VGPR i, lanes 0-15 -> M=i, lanes 16-31 -> M=8+i; N = lane&15
  const int row0 = tm * 16 + half * 8;
#pragma unroll
  for (int t = 0; t < 4; ++t) {
    const int col = (tn0 + t) * 16 + r16;
#pragma unroll
    for (int i = 0; i < 8; ++i)
      C[(size_t)(row0 + i) * ldc + col] = acc[t][i];
  }
}

// ---------------------------------------------------------------------------
// Attention coefficients: one wave per (node, head).
//   es[n,h] = sum_c h[n, h*C + c] * a_s[h,c];  ed analogous.
// ---------------------------------------------------------------------------
__global__ void attn_coef_kernel(const float* __restrict__ h,
                                 const float* __restrict__ a_s,
                                 const float* __restrict__ a_d,
                                 float* __restrict__ es, float* __restrict__ ed,
                                 int N, int H, int C, int ld) {
  int wid  = (int)(((size_t)blockIdx.x * blockDim.x + threadIdx.x) >> 5);
  int lane = threadIdx.x & 31;
  if (wid >= N * H) return;
  int n = wid / H, hh = wid - n * H;
  const float* hp  = h + (size_t)n * ld + hh * C;
  const float* asp = a_s + hh * C;
  const float* adp = a_d + hh * C;
  float s = 0.f, d = 0.f;
  for (int c = lane; c < C; c += 32) {
    float v = hp[c];
    s += v * asp[c];
    d += v * adp[c];
  }
#pragma unroll
  for (int o = 16; o > 0; o >>= 1) {
    s += __shfl_xor(s, o, 32);
    d += __shfl_xor(d, o, 32);
  }
  if (lane == 0) { es[wid] = s; ed[wid] = d; }
}

// Order-preserving float <-> uint encoding for atomicMax on floats.
__device__ __forceinline__ unsigned enc_f32(float v) {
  unsigned u = __float_as_uint(v);
  return (u & 0x80000000u) ? ~u : (u | 0x80000000u);
}
__device__ __forceinline__ float dec_f32(unsigned k) {
  unsigned u = (k & 0x80000000u) ? (k ^ 0x80000000u) : ~k;
  return __uint_as_float(u);
}
#define ENC_NEG_INF 0x007FFFFFu   // enc(-inf)

__device__ __forceinline__ float leaky02(float v) {
  return v > 0.f ? v : 0.2f * v;
}

// Segment max over incoming edges (self-loops appended for e >= E).
__global__ void edge_max_kernel(const int* __restrict__ ei, int E, int N,
                                const float* __restrict__ es,
                                const float* __restrict__ ed,
                                unsigned* __restrict__ menc, int H) {
  size_t t = (size_t)blockIdx.x * blockDim.x + threadIdx.x;
  size_t tot = (size_t)(E + N) * H;
  if (t >= tot) return;
  int e  = (int)(t / H);
  int hh = (int)(t - (size_t)e * H);
  int s = (e < E) ? ei[e]     : (e - E);
  int d = (e < E) ? ei[E + e] : (e - E);
  float v = leaky02(es[s * H + hh] + ed[d * H + hh]);
  atomicMax(&menc[d * H + hh], enc_f32(v));
}

// exp(e - m[dst]) + segment sum of denominators.
__global__ void edge_exp_kernel(const int* __restrict__ ei, int E, int N,
                                const float* __restrict__ es,
                                const float* __restrict__ ed,
                                const unsigned* __restrict__ menc,
                                float* __restrict__ alpha,
                                float* __restrict__ denom, int H) {
  size_t t = (size_t)blockIdx.x * blockDim.x + threadIdx.x;
  size_t tot = (size_t)(E + N) * H;
  if (t >= tot) return;
  int e  = (int)(t / H);
  int hh = (int)(t - (size_t)e * H);
  int s = (e < E) ? ei[e]     : (e - E);
  int d = (e < E) ? ei[E + e] : (e - E);
  float v = leaky02(es[s * H + hh] + ed[d * H + hh]);
  float m = dec_f32(menc[d * H + hh]);
  float x = __expf(v - m);
  alpha[(size_t)e * H + hh] = x;
  atomicAdd(&denom[d * H + hh], x);
}

// out[dst, :] += h[src, :] * alpha_norm — one block (256 thr) per edge.
// The gathered source row is staged through LDS with CDNA5 async loads
// (global_load_async_to_lds_b128 + s_wait_asynccnt), then scattered with
// f32 atomics (resolved in L2). Loops over the padded width `ld`; columns
// >= H*C are exact zeros (zero-padded weights), head index clamped.
__global__ void edge_aggregate_kernel(const int* __restrict__ ei, int E, int N,
                                      const float* __restrict__ h,
                                      const float* __restrict__ alpha,
                                      const float* __restrict__ denom,
                                      float* __restrict__ out,
                                      int H, int C, int ld) {
  __shared__ float smem[256 * 4];
  int e = blockIdx.x;
  int s = (e < E) ? ei[e]     : (e - E);
  int d = (e < E) ? ei[E + e] : (e - E);
  const float* srow = h + (size_t)s * ld;
  float* drow = out + (size_t)d * ld;
  const int nchunk = ld >> 2;                // 16-byte chunks per row
  for (int chunk = threadIdx.x; chunk < nchunk; chunk += blockDim.x) {
    const int i0 = chunk << 2;
    // Async stage 16B of the gathered row into this thread's LDS slot.
    unsigned lds_off = (unsigned)(uintptr_t)(&smem[threadIdx.x * 4]);
    unsigned long long gaddr = (unsigned long long)(uintptr_t)(srow + i0);
    asm volatile("global_load_async_to_lds_b128 %0, %1, off"
                 :: "v"(lds_off), "v"(gaddr) : "memory");
    asm volatile("s_wait_asynccnt 0x0" ::: "memory");
#pragma unroll
    for (int j = 0; j < 4; ++j) {
      int i = i0 + j;
      int hh = i / C; if (hh >= H) hh = H - 1;   // padded cols: value is 0
      float a  = alpha[(size_t)e * H + hh];
      float dn = denom[d * H + hh];
      a = a / fmaxf(dn, 1e-16f);
      atomicAdd(&drow[i], smem[threadIdx.x * 4 + j] * a);
    }
  }
}

// out = elu(agg + bias [+ residual])
__global__ void ep_elu_bias_kernel(const float* __restrict__ agg,
                                   const float* __restrict__ bias,
                                   const float* __restrict__ res,
                                   float* __restrict__ out, int N, int F) {
  size_t i = (size_t)blockIdx.x * blockDim.x + threadIdx.x;
  if (i >= (size_t)N * F) return;
  int c = (int)(i % (size_t)F);
  float v = agg[i] + bias[c] + (res ? res[i] : 0.f);
  out[i] = v > 0.f ? v : (__expf(v) - 1.f);
}

// out[n,c] = sigmoid(mean_h agg[n, h*C + c] + b[c])
__global__ void ep_final_kernel(const float* __restrict__ agg,
                                const float* __restrict__ bias,
                                float* __restrict__ out,
                                int N, int H, int C, int ld) {
  size_t i = (size_t)blockIdx.x * blockDim.x + threadIdx.x;
  if (i >= (size_t)N * C) return;
  int n = (int)(i / (size_t)C);
  int c = (int)(i - (size_t)n * C);
  float s = 0.f;
  for (int hh = 0; hh < H; ++hh) s += agg[(size_t)n * ld + hh * C + c];
  s = s / (float)H + bias[c];
  out[i] = 1.f / (1.f + __expf(-s));
}

// ---------------------------------------------------------------------------
// Host-side helpers (stream-only; graph-capture safe)
// ---------------------------------------------------------------------------
static inline size_t cdiv(size_t a, size_t b) { return (a + b - 1) / b; }

static void run_convert(hipStream_t st, const float* src, unsigned short* dst,
                        int srcRows, int rowsOut, int kin, int kout) {
  size_t total = (size_t)rowsOut * kout;
  f2bf_pad_kernel<<<(unsigned)cdiv(total, 256), 256, 0, st>>>(src, dst, srcRows, kin, kout, total);
}

static void run_gemm(hipStream_t st, const unsigned short* A, const unsigned short* B,
                     float* C, int M, int Np, int K) {
  // Np must be a multiple of 64; M-tiles guarded in-kernel.
  dim3 grid((unsigned)(Np / 64), (unsigned)cdiv((size_t)cdiv(M, 16), 8));
  gemm_bf16_wmma_kernel<<<grid, 256, 0, st>>>(A, B, C, M, Np, K, Np);
}

static void run_fill(hipStream_t st, void* p, unsigned val, size_t n) {
  fill_u32_kernel<<<(unsigned)cdiv(n, 256), 256, 0, st>>>((unsigned*)p, val, n);
}

static void run_gat_attention(hipStream_t st, const float* h, int ld, int H, int C,
                              const float* a_s, const float* a_d,
                              const int* ei, int E, int N,
                              float* es, float* ed, unsigned* menc, float* denom,
                              float* alpha, float* agg) {
  size_t waves = (size_t)N * H;
  attn_coef_kernel<<<(unsigned)cdiv(waves * 32, 256), 256, 0, st>>>(h, a_s, a_d, es, ed, N, H, C, ld);
  run_fill(st, menc, ENC_NEG_INF, (size_t)N * H);
  run_fill(st, denom, 0u, (size_t)N * H);
  run_fill(st, agg, 0u, (size_t)N * ld);
  size_t tot = (size_t)(E + N) * H;
  edge_max_kernel<<<(unsigned)cdiv(tot, 256), 256, 0, st>>>(ei, E, N, es, ed, menc, H);
  edge_exp_kernel<<<(unsigned)cdiv(tot, 256), 256, 0, st>>>(ei, E, N, es, ed, menc, alpha, denom, H);
  edge_aggregate_kernel<<<(unsigned)(E + N), 256, 0, st>>>(ei, E, N, h, alpha, denom, agg, H, C, ld);
}

// ---------------------------------------------------------------------------
// Entry point
// ---------------------------------------------------------------------------
extern "C" void kernel_launch(void* const* d_in, const int* in_sizes, int n_in,
                              void* d_out, int out_size, void* d_ws, size_t ws_size,
                              hipStream_t stream) {
  const float* x   = (const float*)d_in[0];
  const int*   ei  = (const int*)  d_in[1];
  const float* W1  = (const float*)d_in[2];
  const float* as1 = (const float*)d_in[3];
  const float* ad1 = (const float*)d_in[4];
  const float* b1  = (const float*)d_in[5];
  const float* W2  = (const float*)d_in[6];
  const float* as2 = (const float*)d_in[7];
  const float* ad2 = (const float*)d_in[8];
  const float* b2  = (const float*)d_in[9];
  const float* W3  = (const float*)d_in[10];
  const float* as3 = (const float*)d_in[11];
  const float* ad3 = (const float*)d_in[12];
  const float* b3  = (const float*)d_in[13];

  const int F0 = 50;
  const int N  = in_sizes[0] / F0;       // 10000 (multiple of 16)
  const int E  = in_sizes[1] / 2;        // 160000
  const int F0p = 64;                    // K padded for WMMA
  const int F1 = 1024, H1 = 4, C1 = 256;
  const int H3 = 6, C3 = 121, F3 = H3 * C3 /*726*/, F3p = 768; // pad to 64-mult

  // Workspace carve-out
  uint8_t* p = (uint8_t*)d_ws;
  auto carve = [&](size_t bytes) -> void* {
    void* r = (void*)p;
    p += (bytes + 255) & ~(size_t)255;
    return r;
  };
  float*          bufA  = (float*)         carve((size_t)N * F1 * 4); // GEMM out (h)
  float*          bufB  = (float*)         carve((size_t)N * F1 * 4); // aggregation
  float*          Hact  = (float*)         carve((size_t)N * F1 * 4); // activated feats
  unsigned short* hb    = (unsigned short*)carve((size_t)N * F1 * 2); // bf16 activations
  unsigned short* Wb    = (unsigned short*)carve((size_t)F1 * F1 * 2);// bf16 weights
  float*          es    = (float*)         carve((size_t)N * H3 * 4);
  float*          ed    = (float*)         carve((size_t)N * H3 * 4);
  unsigned*       menc  = (unsigned*)      carve((size_t)N * H3 * 4);
  float*          denom = (float*)         carve((size_t)N * H3 * 4);
  float*          alpha = (float*)         carve((size_t)(E + N) * H3 * 4);
  (void)ws_size; (void)n_in;

  // ---- Layer 1: h1 = x @ W1^T ; GAT(H=4,C=256) ; elu(+b1) -> Hact ----
  run_convert(stream, x,  hb, N,    N,    F0, F0p);
  run_convert(stream, W1, Wb, F1,   F1,   F0, F0p);
  run_gemm(stream, hb, Wb, bufA, N, F1, F0p);
  run_gat_attention(stream, bufA, F1, H1, C1, as1, ad1, ei, E, N,
                    es, ed, menc, denom, alpha, bufB);
  ep_elu_bias_kernel<<<(unsigned)cdiv((size_t)N * F1, 256), 256, 0, stream>>>(
      bufB, b1, nullptr, Hact, N, F1);

  // ---- Layer 2: h2 = Hact @ W2^T ; GAT ; elu(+b2 + residual) -> Hact ----
  run_convert(stream, Hact, hb, N,  N,  F1, F1);
  run_convert(stream, W2,   Wb, F1, F1, F1, F1);
  run_gemm(stream, hb, Wb, bufA, N, F1, F1);
  run_gat_attention(stream, bufA, F1, H1, C1, as2, ad2, ei, E, N,
                    es, ed, menc, denom, alpha, bufB);
  ep_elu_bias_kernel<<<(unsigned)cdiv((size_t)N * F1, 256), 256, 0, stream>>>(
      bufB, b2, Hact, Hact, N, F1);  // in-place residual + elu

  // ---- Layer 3: h3 = Hact @ W3^T (726 padded to 768) ; GAT(H=6,C=121) ;
  //      sigmoid(mean_heads + b3) -> d_out ----
  run_convert(stream, Hact, hb, N,  N,   F1, F1);
  run_convert(stream, W3,   Wb, F3, F3p, F1, F1);   // zero-pad rows 726..767
  run_gemm(stream, hb, Wb, bufA, N, F3p, F1);
  run_gat_attention(stream, bufA, F3p, H3, C3, as3, ad3, ei, E, N,
                    es, ed, menc, denom, alpha, bufB);
  ep_final_kernel<<<(unsigned)cdiv((size_t)N * C3, 256), 256, 0, stream>>>(
      bufB, b3, (float*)d_out, N, H3, C3, F3p);
}
